// GAT_41850161332533
// MI455X (gfx1250) — matrix operand order repaired and checked by reading, not compiled, not run on hardware
//
#include <hip/hip_runtime.h>
#include <math.h>

#define N_NODES 50000
#define N_EDGES 800000
#define E_TOT   (N_EDGES + N_NODES)
#define IN_CH   128
#define HID     64
#define OUTC    32
#define NEG_SLOPE 0.2f

typedef __attribute__((ext_vector_type(16))) _Float16 v16h;
typedef __attribute__((ext_vector_type(8)))  float    v8f;

// ---------------------------------------------------------------------------
// H[nrows x NC] = X[nrows x K] * W[K x NC]   (f16 WMMA, f32 accumulate)
// One wave owns a 16-row tile; block = 256 threads = 8 waves = 128 rows.
// W staged in LDS as f16, col-major (lw[n*K+k]) so B fragments are contiguous.
// ---------------------------------------------------------------------------
template<int K, int NC>
__global__ __launch_bounds__(256)
void gemm_wmma_f16(const float* __restrict__ X, const float* __restrict__ W,
                   float* __restrict__ H, int nrows)
{
    __shared__ _Float16 lw[NC * K];
    for (int i = threadIdx.x; i < NC * K; i += 256) {
        int n = i / K, k = i % K;
        lw[i] = (_Float16)W[(size_t)k * NC + n];
    }
    __syncthreads();

    const int wave = threadIdx.x >> 5;
    const int lane = threadIdx.x & 31;
    const int row0 = blockIdx.x * 128 + wave * 16;
    if (row0 >= nrows) return;                 // wave-uniform: EXEC stays all-ones

    const int mrow  = lane & 15;
    const int ahalf = (lane < 16) ? 0 : 16;    // A: lanes 16-31 hold K+16..31
    const int bhalf = (lane < 16) ? 0 : 16;    // B: lanes 16-31 hold K+16..31
    const float* xrow = X + (size_t)(row0 + mrow) * K;

    #pragma unroll
    for (int nt = 0; nt < NC / 16; ++nt) {
        v8f acc = {};
        const _Float16* wcol = lw + (size_t)(nt * 16 + mrow) * K + bhalf;
        #pragma unroll
        for (int kb = 0; kb < K; kb += 32) {
            // A fragment (16x32 f16): a[0..15] = K kb+ahalf+0..15
            const float4* xa = reinterpret_cast<const float4*>(xrow + kb + ahalf);
            float4 p0 = xa[0], p1 = xa[1];
            const float4* xb = reinterpret_cast<const float4*>(xrow + kb + 8 + ahalf);
            float4 p2 = xb[0], p3 = xb[1];
            v16h a;
            a[0] = (_Float16)p0.x; a[1] = (_Float16)p0.y; a[2] = (_Float16)p0.z; a[3] = (_Float16)p0.w;
            a[4] = (_Float16)p1.x; a[5] = (_Float16)p1.y; a[6] = (_Float16)p1.z; a[7] = (_Float16)p1.w;
            a[8] = (_Float16)p2.x; a[9] = (_Float16)p2.y; a[10]= (_Float16)p2.z; a[11]= (_Float16)p2.w;
            a[12]= (_Float16)p3.x; a[13]= (_Float16)p3.y; a[14]= (_Float16)p3.z; a[15]= (_Float16)p3.w;
            // B fragment (32x16 f16): 16 contiguous halves per lane from LDS
            v16h b;
            #pragma unroll
            for (int j = 0; j < 16; ++j) b[j] = wcol[kb + j];
            acc = __builtin_amdgcn_wmma_f32_16x16x32_f16(false, a, false, b,
                                                         (short)0, acc, false, false);
        }
        // D layout: VGPR j -> row0+j (lanes 0-15) / row0+j+8 (lanes 16-31), col = nt*16+mrow
        #pragma unroll
        for (int j = 0; j < 8; ++j) {
            int r = row0 + j + ((lane < 16) ? 0 : 8);
            H[(size_t)r * NC + nt * 16 + mrow] = acc[j];
        }
    }
}

// ---------------------------------------------------------------------------
// Per-node attention coefficients: als[n] = h[n,:].a_src,  ald[n] = h[n,:].a_dst
// One wave per node, shfl reduction (wave32).
// ---------------------------------------------------------------------------
__global__ void node_alpha(const float* __restrict__ H, const float* __restrict__ avs,
                           const float* __restrict__ avd, float* __restrict__ als,
                           float* __restrict__ ald, int n, int C)
{
    int gid  = blockIdx.x * blockDim.x + threadIdx.x;
    int node = gid >> 5, lane = gid & 31;
    if (node >= n) return;
    float ss = 0.f, sd = 0.f;
    for (int c = lane; c < C; c += 32) {
        float h = H[(size_t)node * C + c];
        ss += h * avs[c];
        sd += h * avd[c];
    }
    #pragma unroll
    for (int off = 16; off > 0; off >>= 1) {
        ss += __shfl_xor(ss, off, 32);
        sd += __shfl_xor(sd, off, 32);
    }
    if (lane == 0) { als[node] = ss; ald[node] = sd; }
}

__global__ void fill_f32(float* __restrict__ p, float v, int n)
{
    int i = blockIdx.x * blockDim.x + threadIdx.x;
    if (i < n) p[i] = v;
}

// sign-aware monotone encoding -> int atomics implement float atomic max
__device__ __forceinline__ void atomic_max_f32(float* addr, float v)
{
    if (v >= 0.f) atomicMax((int*)addr, __float_as_int(v));
    else          atomicMin((unsigned int*)addr, __float_as_uint(v));
}

__device__ __forceinline__ void edge_endpoints(const int* __restrict__ ei, int e, int& s, int& d)
{
    if (e < N_EDGES) { s = ei[e]; d = ei[N_EDGES + e]; }
    else             { s = d = e - N_EDGES; }        // self-loop tail
}

// Pass 1: e = leaky_relu(as[src]+ad[dst]); segment max into m[dst]
__global__ void edge_max_kernel(const float* __restrict__ als, const float* __restrict__ ald,
                                const int* __restrict__ ei, float* __restrict__ ebuf,
                                float* __restrict__ m, int etot)
{
    int e = blockIdx.x * blockDim.x + threadIdx.x;
    if (e >= etot) return;
    int s, d; edge_endpoints(ei, e, s, d);
    float v = als[s] + ald[d];
    v = (v > 0.f) ? v : NEG_SLOPE * v;
    ebuf[e] = v;
    atomic_max_f32(m + d, v);
}

// Pass 2: p = exp(e - m[dst]); segment sum into ssum[dst]; ebuf := p
__global__ void edge_expsum_kernel(const int* __restrict__ ei, float* __restrict__ ebuf,
                                   const float* __restrict__ m, float* __restrict__ ssum, int etot)
{
    int e = blockIdx.x * blockDim.x + threadIdx.x;
    if (e >= etot) return;
    int s, d; edge_endpoints(ei, e, s, d);
    float p = __expf(ebuf[e] - m[d]);
    ebuf[e] = p;
    atomicAdd(ssum + d, p);
}

// Pass 3: agg[dst,:] += h[src,:] * p / (s[dst]+eps)   (C/4 threads per edge, float4 gathers)
template<int C>
__global__ void edge_scatter_kernel(const float* __restrict__ H, const float* __restrict__ P,
                                    const float* __restrict__ S, const int* __restrict__ ei,
                                    float* __restrict__ AGG, int etot)
{
    constexpr int TPE = C / 4;
    long long t = (long long)blockIdx.x * blockDim.x + threadIdx.x;
    int e = (int)(t / TPE), q = (int)(t % TPE);
    if (e >= etot) return;
    int s, d; edge_endpoints(ei, e, s, d);
    float alpha = P[e] / (S[d] + 1e-16f);
    float4 h = *reinterpret_cast<const float4*>(H + (size_t)s * C + q * 4);
    float* o = AGG + (size_t)d * C + q * 4;
    atomicAdd(o + 0, h.x * alpha);
    atomicAdd(o + 1, h.y * alpha);
    atomicAdd(o + 2, h.z * alpha);
    atomicAdd(o + 3, h.w * alpha);
}

// out = elu(agg + bias), in place (between layers)
__global__ void bias_elu_kernel(float* __restrict__ A, const float* __restrict__ b, int n, int C)
{
    int i = blockIdx.x * blockDim.x + threadIdx.x;
    if (i >= n) return;
    float v = A[i] + b[i % C];
    A[i] = (v > 0.f) ? v : (__expf(v) - 1.f);
}

__global__ void bias_add_kernel(float* __restrict__ A, const float* __restrict__ b, int n, int C)
{
    int i = blockIdx.x * blockDim.x + threadIdx.x;
    if (i < n) A[i] += b[i % C];
}

// ---------------------------------------------------------------------------
extern "C" void kernel_launch(void* const* d_in, const int* in_sizes, int n_in,
                              void* d_out, int out_size, void* d_ws, size_t ws_size,
                              hipStream_t stream)
{
    (void)in_sizes; (void)n_in; (void)out_size; (void)ws_size;
    const float* x   = (const float*)d_in[0];
    const int*   ei  = (const int*)  d_in[1];
    const float* W1  = (const float*)d_in[2];
    const float* as1 = (const float*)d_in[3];
    const float* ad1 = (const float*)d_in[4];
    const float* b1  = (const float*)d_in[5];
    const float* W2  = (const float*)d_in[6];
    const float* as2 = (const float*)d_in[7];
    const float* ad2 = (const float*)d_in[8];
    const float* b2  = (const float*)d_in[9];
    float* out = (float*)d_out;

    float* ws = (float*)d_ws;
    size_t o = 0;
    float* h1   = ws + o; o += (size_t)N_NODES * HID;
    float* agg1 = ws + o; o += (size_t)N_NODES * HID;   // becomes elu(h) in place
    float* h2   = ws + o; o += (size_t)N_NODES * OUTC;
    float* als  = ws + o; o += N_NODES;
    float* ald  = ws + o; o += N_NODES;
    float* m    = ws + o; o += N_NODES;
    float* ssum = ws + o; o += N_NODES;
    float* ebuf = ws + o; o += E_TOT;

    const dim3 blk(256);
    const int ngrid_n   = (N_NODES + 255) / 256;
    const int ngrid_n32 = (N_NODES * 32 + 255) / 256;
    const int ngrid_h   = (N_NODES * HID + 255) / 256;
    const int ngrid_o   = (N_NODES * OUTC + 255) / 256;
    const int egrid     = (E_TOT + 255) / 256;

    // ---------------- Layer 1 ----------------
    gemm_wmma_f16<IN_CH, HID><<<(N_NODES + 127) / 128, blk, 0, stream>>>(x, W1, h1, N_NODES);
    node_alpha<<<ngrid_n32, blk, 0, stream>>>(h1, as1, ad1, als, ald, N_NODES, HID);
    fill_f32<<<ngrid_n, blk, 0, stream>>>(m, -INFINITY, N_NODES);
    fill_f32<<<ngrid_n, blk, 0, stream>>>(ssum, 0.f, N_NODES);
    fill_f32<<<ngrid_h, blk, 0, stream>>>(agg1, 0.f, N_NODES * HID);
    edge_max_kernel   <<<egrid, blk, 0, stream>>>(als, ald, ei, ebuf, m, E_TOT);
    edge_expsum_kernel<<<egrid, blk, 0, stream>>>(ei, ebuf, m, ssum, E_TOT);
    edge_scatter_kernel<HID><<<(int)(((long long)E_TOT * (HID / 4) + 255) / 256), blk, 0, stream>>>(
        h1, ebuf, ssum, ei, agg1, E_TOT);
    bias_elu_kernel<<<ngrid_h, blk, 0, stream>>>(agg1, b1, N_NODES * HID, HID);

    // ---------------- Layer 2 ----------------
    gemm_wmma_f16<HID, OUTC><<<(N_NODES + 127) / 128, blk, 0, stream>>>(agg1, W2, h2, N_NODES);
    node_alpha<<<ngrid_n32, blk, 0, stream>>>(h2, as2, ad2, als, ald, N_NODES, OUTC);
    fill_f32<<<ngrid_n, blk, 0, stream>>>(m, -INFINITY, N_NODES);
    fill_f32<<<ngrid_n, blk, 0, stream>>>(ssum, 0.f, N_NODES);
    fill_f32<<<ngrid_o, blk, 0, stream>>>(out, 0.f, N_NODES * OUTC);
    edge_max_kernel   <<<egrid, blk, 0, stream>>>(als, ald, ei, ebuf, m, E_TOT);
    edge_expsum_kernel<<<egrid, blk, 0, stream>>>(ei, ebuf, m, ssum, E_TOT);
    edge_scatter_kernel<OUTC><<<(int)(((long long)E_TOT * (OUTC / 4) + 255) / 256), blk, 0, stream>>>(
        h2, ebuf, ssum, ei, out, E_TOT);
    bias_add_kernel<<<ngrid_o, blk, 0, stream>>>(out, b2, N_NODES * OUTC, OUTC);
}
